// DynamicKNNEncoder_52682068853182
// MI455X (gfx1250) — matrix operand order repaired and checked by reading, not compiled, not run on hardware
//
#include <hip/hip_runtime.h>
#include <hip/hip_bf16.h>
#include <stdint.h>

// Problem constants (from reference)
#define NPTS  65536
#define NSAMP 16
#define DFEAT 64
#define CIN0  16
#define CX    67      // D + 3
#define CXPAD 68
#define C2    134     // 2*CX  (K of first GEMM)
#define KPAD  160     // C2 padded to multiple of 32
#define EPSV  1e-5f

typedef __attribute__((ext_vector_type(16))) __bf16   v16bf;
typedef __attribute__((ext_vector_type(8)))  float    v8f;
typedef __attribute__((ext_vector_type(4)))  uint32_t su4;
typedef __attribute__((ext_vector_type(8)))  uint32_t su8;

union ABv { v16bf v; uint4 q[2]; };   // 32B WMMA A/B operand
union P8  { __bf16 h[8]; uint4 q; };  // 8 packed bf16
union F8  { float  f[8]; uint4 q[2]; };

// ---------------------------------------------------------------------------
// TDM: DMA a 1-D run of `nelem` bf16 elements from global memory into LDS.
// D# per CDNA5 ISA ch.8: group0 = {count=1, lds_addr, global_addr, type=2},
// group1 = {data_size=2B, tensor_dim0=tile_dim0=nelem (1-D tile)}.
// Issue from one wave; completion via TENSORcnt.
// ---------------------------------------------------------------------------
__device__ __forceinline__ void tdm_load_bf16(uint32_t lds_off, const void* gptr,
                                              uint32_t nelem) {
  const uint64_t ga = (uint64_t)(uintptr_t)gptr;
  su4 g0;
  g0[0] = 1u;                                               // count=1, user mode
  g0[1] = lds_off;                                          // LDS byte address
  g0[2] = (uint32_t)ga;                                     // global_addr[31:0]
  g0[3] = (uint32_t)((ga >> 32) & 0x01FFFFFFu) | (2u << 30);// global_addr hi | type=2
  su8 g1;
  g1[0] = (1u << 16);                                       // data_size=1 -> 2 bytes
  g1[1] = (nelem & 0xFFFFu) << 16;                          // tensor_dim0[15:0]
  g1[2] = (nelem >> 16) | (1u << 16);                       // tensor_dim0[31:16], dim1=1
  g1[3] = (nelem & 0xFFFFu) << 16;                          // tile_dim0 (16-bit)
  g1[4] = 0u;                                               // tile_dim1/2 unused
  g1[5] = nelem;                                            // tensor_dim0_stride lo
  g1[6] = 0u;
  g1[7] = 0u;
  asm volatile("tensor_load_to_lds %0, %1" :: "s"(g0), "s"(g1) : "memory");
}

// Single-instruction lane-xor via ds_swizzle_b32 (group-of-32 mode).
template <int IMM>
__device__ __forceinline__ float swz(float v) {
  return __int_as_float(__builtin_amdgcn_ds_swizzle(__float_as_int(v), IMM));
}
__device__ __forceinline__ float halfsum(float v) {   // sum over 16-lane half
  v += swz<0x041F>(v);   // xor 1
  v += swz<0x081F>(v);   // xor 2
  v += swz<0x101F>(v);   // xor 4
  v += swz<0x201F>(v);   // xor 8
  return v;
}

// ---------------------------------------------------------------------------
// Weight transpose + f32->bf16: src is (K x 64) row-major, dst is [f][Kpad]
// ---------------------------------------------------------------------------
__global__ void k_convw(const float* __restrict__ src, __bf16* __restrict__ dst,
                        int K, int Kpad) {
  int i = blockIdx.x * blockDim.x + threadIdx.x;
  if (i >= 64 * Kpad) return;
  int f = i / Kpad, k = i - f * Kpad;
  dst[f * Kpad + k] = (k < K) ? (__bf16)src[k * 64 + f] : (__bf16)0.0f;
}

// ---------------------------------------------------------------------------
// Q/K projection: Qx[n][0..63]=feats@wq+bq, [64..66]=xyz, [67]=0. Same for Kx.
// ---------------------------------------------------------------------------
__global__ void k_qk(const float* __restrict__ feats, const float* __restrict__ xyz,
                     const float* __restrict__ wq, const float* __restrict__ bq,
                     const float* __restrict__ wk, const float* __restrict__ bk,
                     float* __restrict__ Qx, float* __restrict__ Kx, int cin) {
  const int n = blockIdx.x;
  const int t = threadIdx.x;
  if (t < 64) {
    float q = bq[t], kk = bk[t];
    for (int c = 0; c < cin; ++c) {
      float fv = feats[(size_t)n * cin + c];
      q  += fv * wq[c * 64 + t];
      kk += fv * wk[c * 64 + t];
    }
    Qx[(size_t)n * CXPAD + t] = q;
    Kx[(size_t)n * CXPAD + t] = kk;
  } else if (t < 67) {
    float v = xyz[(size_t)n * 3 + (t - 64)];
    Qx[(size_t)n * CXPAD + t] = v;
    Kx[(size_t)n * CXPAD + t] = v;
  } else if (t == 67) {
    Qx[(size_t)n * CXPAD + 67] = 0.f;
    Kx[(size_t)n * CXPAD + 67] = 0.f;
  }
}

// ---------------------------------------------------------------------------
// Fused per-point pipeline. 4 waves / block, 1 point / wave.
// STAGE 0: h1 = w1^T @ x                -> block-partial sum/sumsq of h1
// STAGE 1: ... BN1+ReLU, h2 = w2^T @ a1 -> block-partial sum/sumsq of h2
// STAGE 2: ... BN2+ReLU, max over s     -> pooled[n][64] (bf16)
// Weights via TDM; stats via LDS scratch that aliases the dead x^T tile.
// ---------------------------------------------------------------------------
template <int STAGE>
__global__ __launch_bounds__(128)
void k_pass(const float* __restrict__ Qx, const float* __restrict__ Kx,
            const int* __restrict__ knn_idx, const uint8_t* __restrict__ knn_empty,
            const __bf16* __restrict__ w1T, const __bf16* __restrict__ w2T,
            const float* __restrict__ scale1, const float* __restrict__ shift1,
            const float* __restrict__ scale2, const float* __restrict__ shift2,
            float* __restrict__ psum, float* __restrict__ psq,
            __bf16* __restrict__ pooled) {
  __shared__ __align__(16) __bf16 sW1[64][KPAD];        // 20.0 KB (TDM-filled)
  __shared__ __align__(16) __bf16 sW2[64][64];          //  8.0 KB (TDM-filled)
  __shared__ __align__(16) union {
    __bf16 xt[4][NSAMP][KPAD];                          // 20.0 KB  x^T per point
    float  h [4][NSAMP][64];                            // 16.0 KB  stat scratch
  } sX;
  __shared__ __align__(16) __bf16 sA1[4][NSAMP][64];    //  8.0 KB  a1^T per point
  __shared__ float sS1[64], sH1[64], sS2[64], sH2[64];

  const int tid  = threadIdx.x;
  const int w    = tid >> 5;
  const int lane = tid & 31;
  const int half = lane >> 4;     // 0: lanes 0-15, 1: lanes 16-31
  const int l16  = lane & 15;
  const int n    = blockIdx.x * 4 + w;

  // Wave 0: async DMA of both transposed weight matrices into LDS via TDM.
  if (w == 0) {
    tdm_load_bf16((uint32_t)(uintptr_t)&sW1[0][0], w1T, 64u * KPAD);
    tdm_load_bf16((uint32_t)(uintptr_t)&sW2[0][0], w2T, 64u * 64u);
    __builtin_amdgcn_s_wait_tensorcnt(0);
  }
  if (tid < 64) {
    if (STAGE >= 1) { sS1[tid] = scale1[tid]; sH1[tid] = shift1[tid]; }
    if (STAGE >= 2) { sS2[tid] = scale2[tid]; sH2[tid] = shift2[tid]; }
  }

  // Build x^T in LDS for this wave's point: x[k][s], bf16, zero-padded K
  {
    const int  s     = l16;
    const int  idx   = knn_idx[(size_t)n * NSAMP + s];
    const bool empty = knn_empty[n] != 0;
    const float* krow = Kx + (size_t)idx * CXPAD;
    const float* qrow = Qx + (size_t)n * CXPAD;
    __builtin_prefetch(krow, 0, 1);                 // global_prefetch path
    for (int c = half; c < CX; c += 2) {
      float qv = qrow[c];
      float kv = empty ? 0.f : (krow[c] - qv);
      if (empty) qv = 0.f;
      sX.xt[w][s][c]      = (__bf16)kv;
      sX.xt[w][s][CX + c] = (__bf16)qv;
    }
    for (int k = C2 + half; k < KPAD; k += 2) sX.xt[w][s][k] = (__bf16)0.0f;
  }
  __syncthreads();

  // ---- Stage-1 GEMM: h1(f,s) = sum_k w1T[f][k] * xT[s][k], 4 f-tiles, K=160
  v8f acc[4] = {};
  #pragma unroll
  for (int kc = 0; kc < 5; ++kc) {
    const int k0 = kc * 32;
    ABv b;                                  // B: col N=l16(sample), K {.,.+16}
    const int kB = k0 + half * 8;
    b.q[0] = *(const uint4*)&sX.xt[w][l16][kB];
    b.q[1] = *(const uint4*)&sX.xt[w][l16][kB + 16];
    #pragma unroll
    for (int ft = 0; ft < 4; ++ft) {        // A: row M=l16(feature), K {.,.+16}
      ABv a;
      const int kA = k0 + half * 8;
      a.q[0] = *(const uint4*)&sW1[ft * 16 + l16][kA];
      a.q[1] = *(const uint4*)&sW1[ft * 16 + l16][kA + 16];
      acc[ft] = __builtin_amdgcn_wmma_f32_16x16x32_bf16(
          false, a.v, false, b.v, (short)0, acc[ft], false, false);
    }
  }

  if (STAGE == 0) {
    __syncthreads();                 // all waves done reading x^T (alias below)
    #pragma unroll
    for (int ft = 0; ft < 4; ++ft) { // dump C tile to scratch: h[pt][s][f]
      F8 o;
      const int fb = ft * 16 + half * 8;
      #pragma unroll
      for (int r = 0; r < 8; ++r) o.f[r] = acc[ft][r];
      *(uint4*)&sX.h[w][l16][fb]     = o.q[0];
      *(uint4*)&sX.h[w][l16][fb + 4] = o.q[1];
    }
    __syncthreads();
    if (tid < 64) {                  // one feature per thread, bank-conflict-free
      float s = 0.f, q = 0.f;
      for (int pt = 0; pt < 4; ++pt)
        #pragma unroll
        for (int ss = 0; ss < NSAMP; ++ss) {
          float v = sX.h[pt][ss][tid];
          s += v; q += v * v;
        }
      psum[(size_t)blockIdx.x * 64 + tid] = s;
      psq [(size_t)blockIdx.x * 64 + tid] = q;
    }
    return;
  }

  // ---- BN1 + ReLU -> a1^T (bf16) in LDS: a1T[s][f]
  #pragma unroll
  for (int ft = 0; ft < 4; ++ft) {
    P8 p;
    const int fb = ft * 16 + half * 8;
    #pragma unroll
    for (int r = 0; r < 8; ++r) {
      float v = acc[ft][r] * sS1[fb + r] + sH1[fb + r];
      p.h[r] = (__bf16)fmaxf(v, 0.f);
    }
    *(uint4*)&sA1[w][l16][fb] = p.q;       // same-wave LDS RAW, compiler waits
  }

  // ---- Stage-2 GEMM: h2(f,s) = sum_k w2T[f][k] * a1T[s][k], K=64
  v8f acc2[4] = {};
  #pragma unroll
  for (int kc = 0; kc < 2; ++kc) {
    const int k0 = kc * 32;
    ABv b;
    const int kB = k0 + half * 8;
    b.q[0] = *(const uint4*)&sA1[w][l16][kB];
    b.q[1] = *(const uint4*)&sA1[w][l16][kB + 16];
    #pragma unroll
    for (int ft = 0; ft < 4; ++ft) {
      ABv a;
      const int kA = k0 + half * 8;
      a.q[0] = *(const uint4*)&sW2[ft * 16 + l16][kA];
      a.q[1] = *(const uint4*)&sW2[ft * 16 + l16][kA + 16];
      acc2[ft] = __builtin_amdgcn_wmma_f32_16x16x32_bf16(
          false, a.v, false, b.v, (short)0, acc2[ft], false, false);
    }
  }

  if (STAGE == 1) {
    __syncthreads();                 // all waves done with x^T / a1 reads
    #pragma unroll
    for (int ft = 0; ft < 4; ++ft) {
      F8 o;
      const int fb = ft * 16 + half * 8;
      #pragma unroll
      for (int r = 0; r < 8; ++r) o.f[r] = acc2[ft][r];
      *(uint4*)&sX.h[w][l16][fb]     = o.q[0];
      *(uint4*)&sX.h[w][l16][fb + 4] = o.q[1];
    }
    __syncthreads();
    if (tid < 64) {
      float s = 0.f, q = 0.f;
      for (int pt = 0; pt < 4; ++pt)
        #pragma unroll
        for (int ss = 0; ss < NSAMP; ++ss) {
          float v = sX.h[pt][ss][tid];
          s += v; q += v * v;
        }
      psum[(size_t)blockIdx.x * 64 + tid] = s;
      psq [(size_t)blockIdx.x * 64 + tid] = q;
    }
    return;
  }

  // ---- STAGE 2: BN2 + ReLU -> scratch, then max over samples -> pooled (bf16)
  __syncthreads();                   // all waves done with x^T reads (alias)
  #pragma unroll
  for (int ft = 0; ft < 4; ++ft) {
    F8 o;
    const int fb = ft * 16 + half * 8;
    #pragma unroll
    for (int r = 0; r < 8; ++r) {
      float v = acc2[ft][r] * sS2[fb + r] + sH2[fb + r];
      o.f[r] = fmaxf(v, 0.f);
    }
    *(uint4*)&sX.h[w][l16][fb]     = o.q[0];
    *(uint4*)&sX.h[w][l16][fb + 4] = o.q[1];
  }
  __syncthreads();
  for (int i = tid; i < 256; i += 128) {   // (point, feature) pairs
    const int pt = i >> 6, f = i & 63;
    float m = sX.h[pt][0][f];
    #pragma unroll
    for (int ss = 1; ss < NSAMP; ++ss) m = fmaxf(m, sX.h[pt][ss][f]);
    pooled[(size_t)(blockIdx.x * 4 + pt) * 64 + f] = (__bf16)m;
  }
}

// ---------------------------------------------------------------------------
// Output GEMM: ho(N x 64) = pooled(N x 64) @ wo(64 x 64), + BN3 partials.
// 8 waves / block, 16 points / wave. Weight tile via TDM.
// ---------------------------------------------------------------------------
__global__ __launch_bounds__(256)
void k_out(const __bf16* __restrict__ pooled, const __bf16* __restrict__ woT,
           float* __restrict__ ho, float* __restrict__ psum, float* __restrict__ psq) {
  __shared__ __align__(16) __bf16 sWo[64][64];
  __shared__ float sSum[64], sSq[64];
  const int tid = threadIdx.x, w = tid >> 5, lane = tid & 31;
  const int half = lane >> 4, l16 = lane & 15;
  if (w == 0) {
    tdm_load_bf16((uint32_t)(uintptr_t)&sWo[0][0], woT, 64u * 64u);
    __builtin_amdgcn_s_wait_tensorcnt(0);
  }
  if (tid < 64) { sSum[tid] = 0.f; sSq[tid] = 0.f; }
  __syncthreads();

  const int p = (blockIdx.x * 8 + w) * 16 + l16;   // this lane's point (B column)
  v8f acc[4] = {};
  #pragma unroll
  for (int kc = 0; kc < 2; ++kc) {
    const int k0 = kc * 32;
    ABv b;
    const int kB = k0 + half * 8;
    b.q[0] = *(const uint4*)&pooled[(size_t)p * 64 + kB];
    b.q[1] = *(const uint4*)&pooled[(size_t)p * 64 + kB + 16];
    #pragma unroll
    for (int ft = 0; ft < 4; ++ft) {
      ABv a;
      const int kA = k0 + half * 8;
      a.q[0] = *(const uint4*)&sWo[ft * 16 + l16][kA];
      a.q[1] = *(const uint4*)&sWo[ft * 16 + l16][kA + 16];
      acc[ft] = __builtin_amdgcn_wmma_f32_16x16x32_bf16(
          false, a.v, false, b.v, (short)0, acc[ft], false, false);
    }
  }
  #pragma unroll
  for (int ft = 0; ft < 4; ++ft) {
    F8 o;
    const int fb = ft * 16 + half * 8;
    #pragma unroll
    for (int r = 0; r < 8; ++r) {
      float v  = acc[ft][r];
      o.f[r] = v;
      float s1 = halfsum(v);
      float s2 = halfsum(v * v);
      if (l16 == 0) { atomicAdd(&sSum[fb + r], s1); atomicAdd(&sSq[fb + r], s2); }
    }
    *(uint4*)&ho[(size_t)p * 64 + fb]     = o.q[0];
    *(uint4*)&ho[(size_t)p * 64 + fb + 4] = o.q[1];
  }
  __syncthreads();
  if (tid < 64) {
    psum[(size_t)blockIdx.x * 64 + tid] = sSum[tid];
    psq [(size_t)blockIdx.x * 64 + tid] = sSq[tid];
  }
}

// ---------------------------------------------------------------------------
// Reduce per-block partials -> BN scale/shift (64 features, 64 threads).
// ---------------------------------------------------------------------------
__global__ void k_bnstats(const float* __restrict__ psum, const float* __restrict__ psq,
                          int nblocks, float count,
                          const float* __restrict__ g, const float* __restrict__ b,
                          float* __restrict__ scale, float* __restrict__ shift) {
  const int f = threadIdx.x;
  float s = 0.f, q = 0.f;
  for (int i = 0; i < nblocks; ++i) { s += psum[(size_t)i * 64 + f]; q += psq[(size_t)i * 64 + f]; }
  const float m   = s / count;
  const float var = fmaxf(q / count - m * m, 0.f);
  const float sc  = g[f] * rsqrtf(var + EPSV);
  scale[f] = sc;
  shift[f] = b[f] - m * sc;
}

// BN3 apply + ReLU
__global__ void k_final(const float* __restrict__ ho, const float* __restrict__ scale,
                        const float* __restrict__ shift, float* __restrict__ out) {
  const int i = blockIdx.x * blockDim.x + threadIdx.x;
  const int f = i & 63;
  out[i] = fmaxf(ho[i] * scale[f] + shift[f], 0.f);
}

// ---------------------------------------------------------------------------
extern "C" void kernel_launch(void* const* d_in, const int* in_sizes, int n_in,
                              void* d_out, int out_size, void* d_ws, size_t ws_size,
                              hipStream_t stream) {
  (void)in_sizes; (void)n_in; (void)out_size; (void)ws_size;
  const float*   xyz    = (const float*)d_in[0];
  const float*   feats0 = (const float*)d_in[1];
  const int*     kidx   = (const int*)d_in[2];
  const uint8_t* kempty = (const uint8_t*)d_in[3];

  char* ws = (char*)d_ws;
  size_t off = 0;
  auto alloc = [&](size_t bytes) -> void* {
    void* p = ws + off;
    off = (off + bytes + 255) & ~(size_t)255;
    return p;
  };
  float*  Qx     = (float*) alloc((size_t)NPTS * CXPAD * 4);
  float*  Kx     = (float*) alloc((size_t)NPTS * CXPAD * 4);
  float*  fmid   = (float*) alloc((size_t)NPTS * 64 * 4);
  float*  ho     = (float*) alloc((size_t)NPTS * 64 * 4);
  __bf16* pooled = (__bf16*)alloc((size_t)NPTS * 64 * 2);
  __bf16* w1T    = (__bf16*)alloc((size_t)64 * KPAD * 2);
  __bf16* w2T    = (__bf16*)alloc((size_t)64 * 64 * 2);
  __bf16* woT    = (__bf16*)alloc((size_t)64 * 64 * 2);
  float*  psum   = (float*) alloc((size_t)(NPTS / 4) * 64 * 4);
  float*  psq    = (float*) alloc((size_t)(NPTS / 4) * 64 * 4);
  float*  sc1 = (float*)alloc(64 * 4); float* sh1 = (float*)alloc(64 * 4);
  float*  sc2 = (float*)alloc(64 * 4); float* sh2 = (float*)alloc(64 * 4);
  float*  sc3 = (float*)alloc(64 * 4); float* sh3 = (float*)alloc(64 * 4);

  const float* feats = feats0;
  int cin = CIN0;
  for (int layer = 0; layer < 2; ++layer) {
    const int b = 4 + layer * 13;
    const float* wq = (const float*)d_in[b + 0];
    const float* bq = (const float*)d_in[b + 1];
    const float* wk = (const float*)d_in[b + 2];
    const float* bk = (const float*)d_in[b + 3];
    const float* w1 = (const float*)d_in[b + 4];
    const float* g1 = (const float*)d_in[b + 5];
    const float* b1 = (const float*)d_in[b + 6];
    const float* w2 = (const float*)d_in[b + 7];
    const float* g2 = (const float*)d_in[b + 8];
    const float* b2 = (const float*)d_in[b + 9];
    const float* wo = (const float*)d_in[b + 10];
    const float* go = (const float*)d_in[b + 11];
    const float* bo = (const float*)d_in[b + 12];

    k_convw<<<(64 * KPAD + 255) / 256, 256, 0, stream>>>(w1, w1T, C2, KPAD);
    k_convw<<<(64 * 64 + 255) / 256, 256, 0, stream>>>(w2, w2T, 64, 64);
    k_convw<<<(64 * 64 + 255) / 256, 256, 0, stream>>>(wo, woT, 64, 64);
    k_qk<<<NPTS, 96, 0, stream>>>(feats, xyz, wq, bq, wk, bk, Qx, Kx, cin);

    k_pass<0><<<NPTS / 4, 128, 0, stream>>>(Qx, Kx, kidx, kempty, w1T, w2T,
                                            sc1, sh1, sc2, sh2, psum, psq, pooled);
    k_bnstats<<<1, 64, 0, stream>>>(psum, psq, NPTS / 4, (float)((size_t)NPTS * NSAMP),
                                    g1, b1, sc1, sh1);
    k_pass<1><<<NPTS / 4, 128, 0, stream>>>(Qx, Kx, kidx, kempty, w1T, w2T,
                                            sc1, sh1, sc2, sh2, psum, psq, pooled);
    k_bnstats<<<1, 64, 0, stream>>>(psum, psq, NPTS / 4, (float)((size_t)NPTS * NSAMP),
                                    g2, b2, sc2, sh2);
    k_pass<2><<<NPTS / 4, 128, 0, stream>>>(Qx, Kx, kidx, kempty, w1T, w2T,
                                            sc1, sh1, sc2, sh2, psum, psq, pooled);

    k_out<<<NPTS / 128, 256, 0, stream>>>(pooled, woT, ho, psum, psq);
    k_bnstats<<<1, 64, 0, stream>>>(psum, psq, NPTS / 128, (float)NPTS, go, bo, sc3, sh3);

    float* dst = (layer == 0) ? fmid : (float*)d_out;
    k_final<<<(NPTS * 64) / 256, 256, 0, stream>>>(ho, sc3, sh3, dst);

    feats = fmid;
    cin = 64;
  }
}
